// MultiHeadAttention_11562051961033
// MI455X (gfx1250) — compile-verified, matrix operand order checked by reading, not run on hardware
//
#include <hip/hip_runtime.h>
#include <hip/hip_bf16.h>

typedef __attribute__((ext_vector_type(16))) _Float16 v16h;
typedef __attribute__((ext_vector_type(8)))  _Float16 v8h;
typedef __attribute__((ext_vector_type(8)))  float    v8f;
typedef __attribute__((ext_vector_type(4)))  float    v4f;
typedef __attribute__((ext_vector_type(4)))  unsigned int u32x4;
typedef __attribute__((ext_vector_type(4)))  int      i32x4;
typedef __attribute__((ext_vector_type(8)))  int      i32x8;

#define B_   32
#define S_   1024
#define HID  512
#define NH   8
#define DH   64
#define NEGF (-4294967295.0f)  /* -(2^32)+1 */

#if defined(__gfx1250__) && __has_builtin(__builtin_amdgcn_tensor_load_to_lds) && \
    __has_builtin(__builtin_amdgcn_s_wait_tensorcnt)
#define USE_TDM 1
#else
#define USE_TDM 0
#endif

union Frag { v16h v; v8h h[2]; _Float16 e[16]; };

// ---------------------------------------------------------------------------
// Kernel 1: padding masks. One wave per row: sign(|sum over 512 features|).
// ---------------------------------------------------------------------------
__global__ __launch_bounds__(256)
void mask_kernel(const float* __restrict__ Xq, const float* __restrict__ Xk,
                 float* __restrict__ qmask, float* __restrict__ kmask)
{
    const int lane = threadIdx.x & 31;
    const int wave = threadIdx.x >> 5;
    const int row  = blockIdx.x * 8 + wave;          // 0..65535
    const bool isQ = row < (B_ * S_);
    const int  r   = row & (B_ * S_ - 1);
    const float* X = isQ ? Xq : Xk;

    float s = 0.f;
    const float* p = X + (size_t)r * HID + lane;
#pragma unroll
    for (int i = 0; i < 16; ++i) s += p[i * 32];
#pragma unroll
    for (int m = 16; m >= 1; m >>= 1) s += __shfl_xor(s, m, 32);

    if (lane == 0) {
        float v = (fabsf(s) != 0.f) ? 1.f : 0.f;
        (isQ ? qmask : kmask)[r] = v;
    }
}

// ---------------------------------------------------------------------------
// Kernel 2: projection GEMM  out = X[32768x512] * W[512x512] + bias, f16 out.
// A tile (128x32 f32) staged into LDS by the Tensor Data Mover (one D# per
// K-step, issued by wave 0, TENSORcnt-synchronized); W tile staged transposed
// as f16. f32->f16 conversion happens during A-fragment assembly.
// MODE 0: row-major out[row][col].  MODE 1: transposed V store VT[h][b][d][s].
// ---------------------------------------------------------------------------
template <int MODE>
__global__ __launch_bounds__(256)
void proj_kernel(const float* __restrict__ X, const float* __restrict__ W,
                 const float* __restrict__ bias, _Float16* __restrict__ out)
{
    __shared__ __align__(16) float     As[128 * 32];  // [m][k] f32 (TDM dest)
    __shared__ __align__(16) _Float16  Bs[64 * 32];   // [n][k] f16 (W transposed)

    const int tid  = threadIdx.x;
    const int lane = tid & 31;
    const int wave = tid >> 5;
    const int m0   = blockIdx.x * 128;
    const int n0   = blockIdx.y * 64;
    const int wm   = (wave & 3) * 32;
    const int wn   = (wave >> 2) * 32;
    const int l15  = lane & 15;
    const int hs   = (lane < 16) ? 0 : 8;             // K-half select per layout
    const int rh   = (lane < 16) ? 0 : 8;             // C-layout row half

    v8f acc[2][2] = {};

    for (int k0 = 0; k0 < HID; k0 += 32) {
#if USE_TDM
        if (wave == 0) {
            // Build Tensor DMA Descriptor (D#) for the 128x32 f32 tile.
            unsigned long long ga = (unsigned long long)(size_t)
                ((const char*)X + ((size_t)m0 * HID + (size_t)k0) * sizeof(float));
            unsigned int ldsA = (unsigned int)(size_t)(void*)As;
            u32x4 g0;
            g0.x = 1u;                                     // count=1 (valid)
            g0.y = ldsA;                                   // lds_addr (bytes)
            g0.z = (unsigned int)(ga & 0xFFFFFFFFull);     // global_addr[31:0]
            g0.w = (unsigned int)((ga >> 32) & 0x01FFFFFFull) | 0x80000000u; // +type=2
            i32x8 g1;
            g1[0] = (int)(2u << 16);                       // data_size = 4 bytes
            g1[1] = (int)(512u << 16);                     // tensor_dim0 = 512 (lo16)
            g1[2] = (int)(32768u << 16);                   // tensor_dim1 = 32768 (lo16)
            g1[3] = (int)(32u << 16);                      // tile_dim0 = 32 (k)
            g1[4] = 128;                                   // tile_dim1 = 128 (rows)
            g1[5] = 512;                                   // tensor_dim0_stride = 512
            g1[6] = 0;
            g1[7] = 0;
            i32x4 z4 = {};
#if __clang_major__ >= 23
            i32x8 z8 = {};
            __builtin_amdgcn_tensor_load_to_lds(g0, g1, z4, z4, z8, 0);
#else
            __builtin_amdgcn_tensor_load_to_lds(g0, g1, z4, z4, 0);
#endif
            __builtin_amdgcn_s_wait_tensorcnt(0);
        }
#else
        // Fallback: vectorized f32 staging of the A tile.
#pragma unroll
        for (int i = 0; i < 4; ++i) {
            int idx4 = tid + i * 256;                 // 4 floats each, 1024 total
            int r    = idx4 >> 3;
            int c4   = (idx4 & 7) * 4;
            *(v4f*)&As[r * 32 + c4] =
                *(const v4f*)(X + (size_t)(m0 + r) * HID + k0 + c4);
        }
#endif
        // Stage W tile 32x64 transposed -> Bs[n][k], f32 -> f16.
#pragma unroll
        for (int i = 0; i < 2; ++i) {
            int idx4 = tid + i * 256;                 // 4 floats each, 512 total
            int kk   = idx4 >> 4;
            int c4   = (idx4 & 15) * 4;
            v4f w = *(const v4f*)(W + (size_t)(k0 + kk) * HID + n0 + c4);
            Bs[(c4 + 0) * 32 + kk] = (_Float16)w.x;
            Bs[(c4 + 1) * 32 + kk] = (_Float16)w.y;
            Bs[(c4 + 2) * 32 + kk] = (_Float16)w.z;
            Bs[(c4 + 3) * 32 + kk] = (_Float16)w.w;
        }
        __syncthreads();

        Frag af[2], bf[2];
#pragma unroll
        for (int i = 0; i < 2; ++i) {
            // A fragment: f32 LDS rows -> packed f16 halves.
            const float* ap = &As[(wm + i * 16 + l15) * 32];
            v4f a0 = *(const v4f*)(ap + hs);
            v4f a1 = *(const v4f*)(ap + hs + 4);
            v4f a2 = *(const v4f*)(ap + hs + 16);
            v4f a3 = *(const v4f*)(ap + hs + 20);
            af[i].e[0]  = (_Float16)a0.x; af[i].e[1]  = (_Float16)a0.y;
            af[i].e[2]  = (_Float16)a0.z; af[i].e[3]  = (_Float16)a0.w;
            af[i].e[4]  = (_Float16)a1.x; af[i].e[5]  = (_Float16)a1.y;
            af[i].e[6]  = (_Float16)a1.z; af[i].e[7]  = (_Float16)a1.w;
            af[i].e[8]  = (_Float16)a2.x; af[i].e[9]  = (_Float16)a2.y;
            af[i].e[10] = (_Float16)a2.z; af[i].e[11] = (_Float16)a2.w;
            af[i].e[12] = (_Float16)a3.x; af[i].e[13] = (_Float16)a3.y;
            af[i].e[14] = (_Float16)a3.z; af[i].e[15] = (_Float16)a3.w;
            const _Float16* bp = &Bs[(wn + i * 16 + l15) * 32];
            bf[i].h[0] = *(const v8h*)(bp + hs);
            bf[i].h[1] = *(const v8h*)(bp + hs + 16);
        }
#pragma unroll
        for (int i = 0; i < 2; ++i)
#pragma unroll
            for (int j = 0; j < 2; ++j)
                acc[i][j] = __builtin_amdgcn_wmma_f32_16x16x32_f16(
                    false, af[i].v, false, bf[j].v, (short)0, acc[i][j], false, false);
        __syncthreads();
    }

    // epilogue: bias add + f16 store
#pragma unroll
    for (int i = 0; i < 2; ++i) {
#pragma unroll
        for (int j = 0; j < 2; ++j) {
            int col = n0 + wn + j * 16 + l15;
            float bvv = bias[col];
            int rowBase = m0 + wm + i * 16 + rh;
#pragma unroll
            for (int r = 0; r < 8; ++r) {
                int row = rowBase + r;
                float val = acc[i][j][r] + bvv;
                if (MODE == 0) {
                    out[(size_t)row * HID + col] = (_Float16)val;
                } else {
                    int b  = row >> 10, pos = row & 1023;
                    int h  = col >> 6,  dd  = col & 63;
                    out[((size_t)((h * B_ + b) * DH + dd) << 10) + pos] = (_Float16)val;
                }
            }
        }
    }
}

// ---------------------------------------------------------------------------
// Kernel 3: causal flash attention. One wave per (head, batch, 16-row q tile).
// ---------------------------------------------------------------------------
__global__ __launch_bounds__(256)
void attn_kernel(const _Float16* __restrict__ Q, const _Float16* __restrict__ K,
                 const _Float16* __restrict__ VT, const float* __restrict__ kmask,
                 const float* __restrict__ qmask, const float* __restrict__ Xq,
                 float* __restrict__ Out)
{
    __shared__ __align__(16) _Float16 pLds[8][16 * 32]; // per-wave P transpose tile

    const int tid  = threadIdx.x;
    const int lane = tid & 31;
    const int wave = tid >> 5;
    const int task = blockIdx.x * 8 + wave;   // 16384 tasks
    const int qt   = task & 63;
    const int b    = (task >> 6) & 31;
    const int h    = task >> 11;
    const int q0   = qt * 16;
    const int l15  = lane & 15;
    const int hs   = (lane < 16) ? 0 : 8;
    const int rh   = (lane < 16) ? 0 : 8;

    // Q fragments (A layout maps directly onto row-major f16 rows)
    Frag aq[2];
    {
        const _Float16* qp = Q + (size_t)(b * S_ + q0 + l15) * HID + h * DH;
#pragma unroll
        for (int i = 0; i < 2; ++i) {
            aq[i].h[0] = *(const v8h*)(qp + i * 32 + hs);
            aq[i].h[1] = *(const v8h*)(qp + i * 32 + hs + 16);
        }
    }

    v8f acc[4] = {};
    float mrow[8], lrow[8];
#pragma unroll
    for (int r = 0; r < 8; ++r) { mrow[r] = -1e30f; lrow[r] = 0.f; }

    const float scale = 0.044194173824159216f;   // 1/sqrt(512)
    const int kend = q0 + 16;                    // causal upper bound (exclusive)
    const _Float16* vbase = VT + (((size_t)((h * B_ + b) * DH)) << 10);

    for (int kc = 0; kc < kend; kc += 32) {
        // K^T fragments: 2 key sub-tiles x 2 d-halves
        Frag bk[2][2];
#pragma unroll
        for (int n = 0; n < 2; ++n) {
            const _Float16* kp = K + (size_t)(b * S_ + kc + n * 16 + l15) * HID + h * DH;
#pragma unroll
            for (int dc = 0; dc < 2; ++dc) {
                bk[n][dc].h[0] = *(const v8h*)(kp + dc * 32 + hs);
                bk[n][dc].h[1] = *(const v8h*)(kp + dc * 32 + hs + 16);
            }
        }
        // Prefetch next causal chunk of K and V^T (global_prefetch_b8).
        if (kc + 32 < kend) {
            __builtin_prefetch(
                (const void*)(K + (size_t)(b * S_ + kc + 32 + l15) * HID + h * DH), 0, 1);
            __builtin_prefetch(
                (const void*)(vbase + (((size_t)l15) << 10) + kc + 32), 0, 1);
        }

        v8f s0 = {}, s1 = {};
        s0 = __builtin_amdgcn_wmma_f32_16x16x32_f16(false, aq[0].v, false, bk[0][0].v, (short)0, s0, false, false);
        s0 = __builtin_amdgcn_wmma_f32_16x16x32_f16(false, aq[1].v, false, bk[0][1].v, (short)0, s0, false, false);
        s1 = __builtin_amdgcn_wmma_f32_16x16x32_f16(false, aq[0].v, false, bk[1][0].v, (short)0, s1, false, false);
        s1 = __builtin_amdgcn_wmma_f32_16x16x32_f16(false, aq[1].v, false, bk[1][1].v, (short)0, s1, false, false);

        const int col0 = kc + l15;
        const int col1 = kc + 16 + l15;
        const float km0 = kmask[b * S_ + col0];
        const float km1 = kmask[b * S_ + col1];

        float p0[8], p1[8];
#pragma unroll
        for (int r = 0; r < 8; ++r) {
            int row = q0 + r + rh;
            float v0 = (col0 <= row && km0 != 0.f) ? s0[r] * scale : NEGF;
            float v1 = (col1 <= row && km1 != 0.f) ? s1[r] * scale : NEGF;
            float cm = fmaxf(v0, v1);
#pragma unroll
            for (int m = 1; m < 16; m <<= 1) cm = fmaxf(cm, __shfl_xor(cm, m, 32));
            float mnew = fmaxf(mrow[r], cm);
            float corr = __expf(mrow[r] - mnew);
            float e0 = __expf(v0 - mnew);
            float e1 = __expf(v1 - mnew);
            float ps = e0 + e1;
#pragma unroll
            for (int m = 1; m < 16; m <<= 1) ps += __shfl_xor(ps, m, 32);
            lrow[r] = lrow[r] * corr + ps;
            mrow[r] = mnew;
#pragma unroll
            for (int t = 0; t < 4; ++t) acc[t][r] *= corr;
            p0[r] = e0; p1[r] = e1;
        }

        // C-layout -> A-layout transpose of P through per-wave LDS tile
        _Float16* pw = pLds[wave];
#pragma unroll
        for (int r = 0; r < 8; ++r) {
            int rq = r + rh;
            pw[rq * 32 + l15]      = (_Float16)p0[r];
            pw[rq * 32 + 16 + l15] = (_Float16)p1[r];
        }
        asm volatile("s_wait_dscnt 0" ::: "memory");
        Frag pa;
        const _Float16* pr = pw + l15 * 32;
        pa.h[0] = *(const v8h*)(pr + hs);
        pa.h[1] = *(const v8h*)(pr + hs + 16);

        // PV: 4 output column sub-tiles against VT (contiguous along key pos)
#pragma unroll
        for (int n = 0; n < 4; ++n) {
            Frag bv;
            const _Float16* vp = vbase + (((size_t)(n * 16 + l15)) << 10) + kc;
            bv.h[0] = *(const v8h*)(vp + hs);
            bv.h[1] = *(const v8h*)(vp + hs + 16);
            acc[n] = __builtin_amdgcn_wmma_f32_16x16x32_f16(
                false, pa.v, false, bv.v, (short)0, acc[n], false, false);
        }
    }

    // epilogue: normalize, query mask, residual add
#pragma unroll
    for (int r = 0; r < 8; ++r) {
        int row = q0 + r + rh;
        float inv = (lrow[r] > 0.f) ? 1.f / lrow[r] : 0.f;
        float f = inv * qmask[b * S_ + row];
#pragma unroll
        for (int n = 0; n < 4; ++n) {
            int col = h * DH + n * 16 + l15;
            size_t o = (size_t)(b * S_ + row) * HID + col;
            Out[o] = acc[n][r] * f + Xq[o];
        }
    }
}

// ---------------------------------------------------------------------------
extern "C" void kernel_launch(void* const* d_in, const int* in_sizes, int n_in,
                              void* d_out, int out_size, void* d_ws, size_t ws_size,
                              hipStream_t stream)
{
    const float* queries = (const float*)d_in[0];
    const float* keys    = (const float*)d_in[1];
    const float* Wq      = (const float*)d_in[2];
    const float* bq      = (const float*)d_in[3];
    const float* Wk      = (const float*)d_in[4];
    const float* bk      = (const float*)d_in[5];
    const float* Wv      = (const float*)d_in[6];
    const float* bv      = (const float*)d_in[7];
    float* out = (float*)d_out;

    char* ws = (char*)d_ws;
    const size_t qkvBytes = (size_t)B_ * S_ * HID * sizeof(_Float16); // 32 MB each
    _Float16* qF = (_Float16*)(ws);
    _Float16* kF = (_Float16*)(ws + qkvBytes);
    _Float16* vT = (_Float16*)(ws + 2 * qkvBytes);
    float* kmaskP = (float*)(ws + 3 * qkvBytes);
    float* qmaskP = (float*)(ws + 3 * qkvBytes + (size_t)B_ * S_ * sizeof(float));

    mask_kernel<<<8192, 256, 0, stream>>>(queries, keys, qmaskP, kmaskP);
    proj_kernel<0><<<dim3(256, 8), 256, 0, stream>>>(queries, Wq, bq, qF);
    proj_kernel<0><<<dim3(256, 8), 256, 0, stream>>>(keys,    Wk, bk, kF);
    proj_kernel<1><<<dim3(256, 8), 256, 0, stream>>>(keys,    Wv, bv, vT);
    attn_kernel<<<2048, 256, 0, stream>>>(qF, kF, vT, kmaskP, qmaskP, queries, out);
}
